// AttnBlock_26388279067259
// MI455X (gfx1250) — compile-verified, matrix operand order checked by reading, not run on hardware
//
#include <hip/hip_runtime.h>
#include <hip/hip_bf16.h>
#include <math.h>

// ---------------------------------------------------------------------------
// CDNA5 (gfx1250, wave32) attention block:
//   GN -> qkv (1x1 conv) -> flash softmax(q^T k / sqrt(C)) v -> proj + residual
// All GEMMs use v_wmma_f32_16x16x32_bf16 (f32 accumulate).
// Flash loop prefetches the next K/V block (global_prefetch_b8).
// ---------------------------------------------------------------------------

typedef __attribute__((ext_vector_type(16))) __bf16 bf16x16;
typedef __attribute__((ext_vector_type(8)))  __bf16 bf16x8;
typedef __attribute__((ext_vector_type(8)))  float  f32x8;

#define WMMA_BF16(A, B, Cacc) \
  __builtin_amdgcn_wmma_f32_16x16x32_bf16(false, (A), false, (B), (short)0, (Cacc), false, false)

static constexpr int Bn  = 4;     // batch
static constexpr int Cd  = 512;   // channels
static constexpr int Nd  = 4096;  // H*W
static constexpr int Cpg = 16;    // channels per group (512/32)

static __device__ __forceinline__ __bf16 f2bf(float f) {
  unsigned int u = __builtin_bit_cast(unsigned int, f);
  u += 0x7FFFu + ((u >> 16) & 1u);              // round-to-nearest-even
  unsigned short s = (unsigned short)(u >> 16);
  return __builtin_bit_cast(__bf16, s);
}

// Load one 16x32 bf16 WMMA operand fragment (A layout; B uses the same lane
// map indexed by column). Lane base pointer already includes row*ld + hi8.
// Elements 0..7 = K{+0..+7}, elements 8..15 = K{+16..+23}.
static __device__ __forceinline__ bf16x16 load_frag(const __bf16* p) {
  bf16x8 lo = *(const bf16x8*)p;
  bf16x8 hi = *(const bf16x8*)(p + 16);
  bf16x16 r;
#pragma unroll
  for (int i = 0; i < 8; ++i) { r[i] = lo[i]; r[i + 8] = hi[i]; }
  return r;
}

// ---------------------------------------------------------------------------
// 1) fp32 -> bf16 weight conversion (wq|wk|wv|wo packed contiguously)
// ---------------------------------------------------------------------------
__global__ __launch_bounds__(256) void k_convert_w(const float* __restrict__ wq,
                                                   const float* __restrict__ wk,
                                                   const float* __restrict__ wv,
                                                   const float* __restrict__ wo,
                                                   __bf16* __restrict__ dst) {
  int idx = blockIdx.x * 256 + threadIdx.x;     // 0 .. 4*512*512-1
  int which = idx >> 18;
  int off   = idx & 262143;
  const float* src = (which == 0) ? wq : (which == 1) ? wk : (which == 2) ? wv : wo;
  dst[idx] = f2bf(src[off]);
}

// ---------------------------------------------------------------------------
// 2) GroupNorm over (C/G, H, W); writes h transposed to [B, N, C] bf16 so the
//    QKV GEMM A-fragments (row n, contiguous c) are 16B vector loads.
// ---------------------------------------------------------------------------
__global__ __launch_bounds__(256) void k_groupnorm(const float* __restrict__ x,
                                                   const float* __restrict__ gw,
                                                   const float* __restrict__ gb,
                                                   __bf16* __restrict__ h_t) {
  const int b = blockIdx.x >> 5;
  const int g = blockIdx.x & 31;
  const int tid = threadIdx.x;
  const float* xg = x + ((size_t)(b * Cd + g * Cpg)) * Nd;

  float s = 0.f, s2 = 0.f;
  for (int i = tid; i < Cpg * Nd; i += 256) {
    float v = xg[i];
    s += v; s2 += v * v;
  }
  __shared__ float r1[256], r2[256];
  r1[tid] = s; r2[tid] = s2;
  __syncthreads();
  for (int st = 128; st > 0; st >>= 1) {
    if (tid < st) { r1[tid] += r1[tid + st]; r2[tid] += r2[tid + st]; }
    __syncthreads();
  }
  const float inv_n = 1.0f / (float)(Cpg * Nd);
  const float mean  = r1[0] * inv_n;
  const float var   = r2[0] * inv_n - mean * mean;
  const float rstd  = rsqrtf(var + 1e-5f);

  for (int i = tid; i < Cpg * Nd; i += 256) {
    int ci = i >> 12, n = i & (Nd - 1);
    int c  = g * Cpg + ci;
    float v = (xg[i] - mean) * rstd * gw[c] + gb[c];
    h_t[((size_t)(b * Nd + n)) * Cd + c] = f2bf(v);
  }
}

// ---------------------------------------------------------------------------
// 3) Fused QKV projection: Q[B,N,C], K[B,N,C] row-major; V stored transposed
//    [B,C,N] so the attention P*V B-fragments are key-contiguous.
//    Grid (N/64, C/64, B), 256 threads = 8 waves; each wave: one 16-row
//    M-tile x two 16-col N-tiles, A-fragment shared across all 3 GEMMs.
// ---------------------------------------------------------------------------
__global__ __launch_bounds__(256) void k_qkv(const __bf16* __restrict__ h_t,
                                             const __bf16* __restrict__ wqb,
                                             const __bf16* __restrict__ wkb,
                                             const __bf16* __restrict__ wvb,
                                             const float* __restrict__ bq,
                                             const float* __restrict__ bk,
                                             const float* __restrict__ bv,
                                             __bf16* __restrict__ Q,
                                             __bf16* __restrict__ Kd,
                                             __bf16* __restrict__ Vt) {
  const int b    = blockIdx.z;
  const int m0   = blockIdx.x * 64;
  const int o0   = blockIdx.y * 64;
  const int w    = threadIdx.x >> 5;
  const int lane = threadIdx.x & 31;
  const int lrow = lane & 15;
  const int hi   = lane >> 4;
  const int hi8  = hi * 8;
  const int mrow  = m0 + (w & 3) * 16;
  const int ocol0 = o0 + (w >> 2) * 32;

  const __bf16* abase = h_t + ((size_t)(b * Nd + mrow + lrow)) * Cd + hi8;
  const __bf16* q0 = wqb + (size_t)(ocol0 + lrow) * Cd + hi8;
  const __bf16* q1 = q0 + 16 * Cd;
  const __bf16* k0 = wkb + (size_t)(ocol0 + lrow) * Cd + hi8;
  const __bf16* k1 = k0 + 16 * Cd;
  const __bf16* v0 = wvb + (size_t)(ocol0 + lrow) * Cd + hi8;
  const __bf16* v1 = v0 + 16 * Cd;

  f32x8 aq0 = {}, aq1 = {}, ak0 = {}, ak1 = {}, av0 = {}, av1 = {};
#pragma unroll 4
  for (int k = 0; k < Cd; k += 32) {
    bf16x16 af = load_frag(abase + k);
    aq0 = WMMA_BF16(af, load_frag(q0 + k), aq0);
    aq1 = WMMA_BF16(af, load_frag(q1 + k), aq1);
    ak0 = WMMA_BF16(af, load_frag(k0 + k), ak0);
    ak1 = WMMA_BF16(af, load_frag(k1 + k), ak1);
    av0 = WMMA_BF16(af, load_frag(v0 + k), av0);
    av1 = WMMA_BF16(af, load_frag(v1 + k), av1);
  }

  // Q / K stores: D-fragment lane (M=r+8*hi, Ncol=lrow) -> [B,N,C] (stride C)
#pragma unroll
  for (int j = 0; j < 2; ++j) {
    const int o = ocol0 + j * 16 + lrow;
    const f32x8& accq = j ? aq1 : aq0;
    const f32x8& acck = j ? ak1 : ak0;
#pragma unroll
    for (int r = 0; r < 8; ++r) {
      size_t n = (size_t)(b * Nd + mrow + 8 * hi + r);
      Q [n * Cd + o] = f2bf(accq[r] + bq[o]);
      Kd[n * Cd + o] = f2bf(acck[r] + bk[o]);
    }
  }
  // V transposed store: rows vary with VGPR index -> 8 contiguous bf16 (b128)
#pragma unroll
  for (int j = 0; j < 2; ++j) {
    const int o = ocol0 + j * 16 + lrow;
    const f32x8& accv = j ? av1 : av0;
    bf16x8 vv;
#pragma unroll
    for (int r = 0; r < 8; ++r) vv[r] = f2bf(accv[r] + bv[o]);
    *(bf16x8*)(Vt + ((size_t)(b * Cd + o)) * Nd + mrow + 8 * hi) = vv;
  }
}

// ---------------------------------------------------------------------------
// 4) Flash attention: grid (N/32, B), 256 threads = 8 waves.
//    WG owns a 32-query tile (two 16-row WMMA tiles -> K/V fragments reused
//    twice in-register, halving L2 traffic); wave w owns channel slice
//    [w*64, w*64+64).  Per 32-key block: partial S (8 WMMAs/wave), partials
//    reduced in LDS, wave 0 runs the online softmax (lane = query row) and
//    publishes P (bf16) + per-row rescale factors, then every wave does
//    O = O*corr + P*V_slice (8 WMMAs/wave).  Next block's K/V lines are
//    prefetched (global_prefetch_b8) at the top of the loop.
// ---------------------------------------------------------------------------
__global__ __launch_bounds__(256) void k_attn(const __bf16* __restrict__ Q,
                                              const __bf16* __restrict__ Kd,
                                              const __bf16* __restrict__ Vt,
                                              __bf16* __restrict__ attnO) {
  const int b     = blockIdx.y;
  const int qbase = blockIdx.x * 32;
  const int tid   = threadIdx.x;
  const int w     = tid >> 5;
  const int lane  = tid & 31;
  const int lrow  = lane & 15;
  const int hi    = lane >> 4;
  const int hi8   = hi * 8;
  const float scale = 0.044194173824159216f; // 512^-0.5

  __shared__ float  Spart[8][32][32];   // per-wave partial scores (32 KB)
  __shared__ float  Sfin[32][32];
  __shared__ __bf16 Pl[32][32];
  __shared__ float  corr_l[32];
  __shared__ float  linv[32];

  // Persistent Q slices: two 16-row tiles x two K=32 chunks.
  bf16x16 qf[2][2];
#pragma unroll
  for (int qt = 0; qt < 2; ++qt)
#pragma unroll
    for (int s = 0; s < 2; ++s)
      qf[qt][s] = load_frag(Q + ((size_t)(b * Nd + qbase + qt * 16 + lrow)) * Cd +
                            w * 64 + s * 32 + hi8);

  f32x8 o_acc[2][4] = {{{}, {}, {}, {}}, {{}, {}, {}, {}}};
  float m_prev = -1e30f, l_run = 0.f;   // live in wave-0 (lane = query row)

  for (int kb = 0; kb < Nd / 32; ++kb) {
    // --- prefetch next key block's K/V lines toward this WGP ---------------
    if (kb + 1 < Nd / 32) {
      __builtin_prefetch(Kd + ((size_t)(b * Nd + (kb + 1) * 32 + lane)) * Cd + w * 64, 0, 3);
      __builtin_prefetch(Vt + ((size_t)(b * Cd + w * 64 + lane)) * Nd + (kb + 1) * 32, 0, 3);
      __builtin_prefetch(Vt + ((size_t)(b * Cd + w * 64 + 32 + lane)) * Nd + (kb + 1) * 32, 0, 3);
    }

    // --- partial S over this wave's 64-channel slice -----------------------
    f32x8 sacc[2][2] = {{{}, {}}, {{}, {}}};
#pragma unroll
    for (int s = 0; s < 2; ++s) {
      bf16x16 kf0 = load_frag(Kd + ((size_t)(b * Nd + kb * 32 + lrow)) * Cd + w * 64 + s * 32 + hi8);
      bf16x16 kf1 = load_frag(Kd + ((size_t)(b * Nd + kb * 32 + 16 + lrow)) * Cd + w * 64 + s * 32 + hi8);
#pragma unroll
      for (int qt = 0; qt < 2; ++qt) {
        sacc[qt][0] = WMMA_BF16(qf[qt][s], kf0, sacc[qt][0]);
        sacc[qt][1] = WMMA_BF16(qf[qt][s], kf1, sacc[qt][1]);
      }
    }
#pragma unroll
    for (int qt = 0; qt < 2; ++qt)
#pragma unroll
      for (int r = 0; r < 8; ++r) {
        Spart[w][qt * 16 + 8 * hi + r][lrow]      = sacc[qt][0][r];
        Spart[w][qt * 16 + 8 * hi + r][16 + lrow] = sacc[qt][1][r];
      }
    __syncthreads();

    // --- cross-wave reduction of the 32x32 score block ----------------------
    for (int e = tid; e < 1024; e += 256) {
      int rr = e >> 5, cc = e & 31;
      float acc = 0.f;
#pragma unroll
      for (int ww = 0; ww < 8; ++ww) acc += Spart[ww][rr][cc];
      Sfin[rr][cc] = acc * scale;
    }
    __syncthreads();

    // --- online softmax (wave 0, one lane per query row) --------------------
    if (tid < 32) {
      float mb = -1e30f;
#pragma unroll 8
      for (int j = 0; j < 32; ++j) mb = fmaxf(mb, Sfin[tid][j]);
      float m_new = fmaxf(m_prev, mb);
      float c0 = __expf(m_prev - m_new);
      float ssum = 0.f;
#pragma unroll 8
      for (int j = 0; j < 32; ++j) {
        float p = __expf(Sfin[tid][j] - m_new);
        Pl[tid][j] = f2bf(p);
        ssum += p;
      }
      l_run = l_run * c0 + ssum;
      corr_l[tid] = c0;
      m_prev = m_new;
    }
    __syncthreads();

    // --- O = O * corr + P x V_slice ----------------------------------------
    float cv[2][8];
#pragma unroll
    for (int qt = 0; qt < 2; ++qt)
#pragma unroll
      for (int r = 0; r < 8; ++r) cv[qt][r] = corr_l[qt * 16 + 8 * hi + r];
    bf16x16 pf[2];
#pragma unroll
    for (int qt = 0; qt < 2; ++qt) pf[qt] = load_frag(&Pl[qt * 16 + lrow][hi8]);
#pragma unroll
    for (int nt = 0; nt < 4; ++nt) {
      bf16x16 vf = load_frag(Vt + ((size_t)(b * Cd + w * 64 + nt * 16 + lrow)) * Nd +
                             kb * 32 + hi8);
#pragma unroll
      for (int qt = 0; qt < 2; ++qt) {
#pragma unroll
        for (int r = 0; r < 8; ++r) o_acc[qt][nt][r] *= cv[qt][r];
        o_acc[qt][nt] = WMMA_BF16(pf[qt], vf, o_acc[qt][nt]);
      }
    }
    __syncthreads();   // protect Spart/Pl/corr before next iteration
  }

  if (tid < 32) linv[tid] = 1.0f / l_run;
  __syncthreads();
  float lv[2][8];
#pragma unroll
  for (int qt = 0; qt < 2; ++qt)
#pragma unroll
    for (int r = 0; r < 8; ++r) lv[qt][r] = linv[qt * 16 + 8 * hi + r];

#pragma unroll
  for (int qt = 0; qt < 2; ++qt)
#pragma unroll
    for (int nt = 0; nt < 4; ++nt)
#pragma unroll
      for (int r = 0; r < 8; ++r)
        attnO[((size_t)(b * Nd + qbase + qt * 16 + 8 * hi + r)) * Cd +
              w * 64 + nt * 16 + lrow] = f2bf(o_acc[qt][nt][r] * lv[qt][r]);
}

// ---------------------------------------------------------------------------
// 5) Output projection + bias + residual (fp32 out, [B, C, N]).
//    M = output channel o, Ncol = spatial n, K = c.
// ---------------------------------------------------------------------------
__global__ __launch_bounds__(256) void k_proj(const __bf16* __restrict__ wob,
                                              const __bf16* __restrict__ attnO,
                                              const float* __restrict__ bo,
                                              const float* __restrict__ res,
                                              float* __restrict__ out) {
  const int b    = blockIdx.z;
  const int n0   = blockIdx.x * 64;
  const int o0   = blockIdx.y * 64;
  const int w    = threadIdx.x >> 5;
  const int lane = threadIdx.x & 31;
  const int lrow = lane & 15;
  const int hi   = lane >> 4;
  const int hi8  = hi * 8;
  const int orow  = o0 + (w & 3) * 16;
  const int ncol0 = n0 + (w >> 2) * 32;

  const __bf16* abase = wob + (size_t)(orow + lrow) * Cd + hi8;
  const __bf16* b0 = attnO + ((size_t)(b * Nd + ncol0 + lrow)) * Cd + hi8;
  const __bf16* b1 = b0 + 16 * Cd;

  f32x8 a0 = {}, a1 = {};
#pragma unroll 4
  for (int k = 0; k < Cd; k += 32) {
    bf16x16 af = load_frag(abase + k);
    a0 = WMMA_BF16(af, load_frag(b0 + k), a0);
    a1 = WMMA_BF16(af, load_frag(b1 + k), a1);
  }

#pragma unroll
  for (int j = 0; j < 2; ++j) {
    const f32x8& acc = j ? a1 : a0;
    const int n = ncol0 + j * 16 + lrow;
#pragma unroll
    for (int r = 0; r < 8; ++r) {
      const int o = orow + 8 * hi + r;
      size_t idx = ((size_t)(b * Cd + o)) * Nd + n;
      out[idx] = acc[r] + bo[o] + res[idx];
    }
  }
}

// ---------------------------------------------------------------------------
extern "C" void kernel_launch(void* const* d_in, const int* in_sizes, int n_in,
                              void* d_out, int out_size, void* d_ws, size_t ws_size,
                              hipStream_t stream) {
  const float* hs  = (const float*)d_in[0];
  const float* gnw = (const float*)d_in[1];
  const float* gnb = (const float*)d_in[2];
  const float* wq  = (const float*)d_in[3];
  const float* bq  = (const float*)d_in[4];
  const float* wk  = (const float*)d_in[5];
  const float* bk  = (const float*)d_in[6];
  const float* wv  = (const float*)d_in[7];
  const float* bv  = (const float*)d_in[8];
  const float* wo  = (const float*)d_in[9];
  const float* bo  = (const float*)d_in[10];
  float* out = (float*)d_out;

  // Workspace layout (bytes):  weights_bf16 [2MB] | h_t [16MB] | Q [16MB] |
  //                            K [16MB] | V_t [16MB] | attnO [16MB]  = 82MB
  const size_t MB = 1ull << 20;
  char* ws = (char*)d_ws;
  __bf16* wAll  = (__bf16*)(ws);
  __bf16* wqb   = wAll;
  __bf16* wkb   = wAll + 1 * 262144;
  __bf16* wvb   = wAll + 2 * 262144;
  __bf16* wob   = wAll + 3 * 262144;
  __bf16* h_t   = (__bf16*)(ws + 2 * MB);
  __bf16* Qb    = (__bf16*)(ws + 18 * MB);
  __bf16* Kb    = (__bf16*)(ws + 34 * MB);
  __bf16* Vtb   = (__bf16*)(ws + 50 * MB);
  __bf16* attnO = (__bf16*)(ws + 66 * MB);

  k_convert_w<<<4096, 256, 0, stream>>>(wq, wk, wv, wo, wAll);
  k_groupnorm<<<Bn * 32, 256, 0, stream>>>(hs, gnw, gnb, h_t);
  k_qkv<<<dim3(Nd / 64, Cd / 64, Bn), 256, 0, stream>>>(h_t, wqb, wkb, wvb,
                                                        bq, bk, bv, Qb, Kb, Vtb);
  k_attn<<<dim3(Nd / 32, Bn), 256, 0, stream>>>(Qb, Kb, Vtb, attnO);
  k_proj<<<dim3(Nd / 64, Cd / 64, Bn), 256, 0, stream>>>(wob, attnO, bo, hs, out);
}
